// Net_screen_9887014715914
// MI455X (gfx1250) — matrix operand order repaired and checked by reading, not compiled, not run on hardware
//
#include <hip/hip_runtime.h>
#include <hip/hip_bf16.h>
#include <stdint.h>

#define NNODES 50000
#define NEDGES 500000
#define DDIM   128
#define GG     64

typedef __attribute__((ext_vector_type(16))) __bf16 v16bf;
typedef __attribute__((ext_vector_type(8)))  float  v8f;

// ---------------------------------------------------------------- helpers
__device__ inline unsigned short f2bf(float f) {
  unsigned int u = __float_as_uint(f);
  unsigned int r = u + 0x7FFFu + ((u >> 16) & 1u);   // round-nearest-even
  return (unsigned short)(r >> 16);
}

__device__ inline void atomicMaxF(float* addr, float val) {
  if (val >= 0.0f) atomicMax((int*)addr, __float_as_int(val));
  else             atomicMin((unsigned int*)addr, __float_as_uint(val));
}

// ---------------------------------------------------------------- utility kernels
__global__ void fill_f32(float* p, float v, int n) {
  int i = blockIdx.x * blockDim.x + threadIdx.x;
  if (i < n) p[i] = v;
}

__global__ void copy_f32(const float* __restrict__ s, float* __restrict__ d, int n) {
  int i = blockIdx.x * blockDim.x + threadIdx.x;
  if (i < n) d[i] = s[i];
}

// fp32 [rows x K] -> bf16 [rows x Kpad], zero padded K..Kpad
__global__ void cvt_pad(const float* __restrict__ src, unsigned short* __restrict__ dst,
                        int K, int Kpad, int n_total) {
  int idx = blockIdx.x * blockDim.x + threadIdx.x;
  if (idx >= n_total) return;
  int r = idx / Kpad, k = idx - r * Kpad;
  dst[idx] = (k < K) ? f2bf(src[(size_t)r * K + k]) : (unsigned short)0;
}

// Pack fp32 weight [K x 128] into WMMA B-fragment order (bf16), zero-pad K->Kpad.
// Layout: dst[((kt*NT_total + nt_off + nt)*32 + lane)*16 + i]
//   lane<16 : N = nt*16 + lane,     K = kt*32 + i
//   lane>=16: N = nt*16 + lane-16,  K = kt*32 + 16 + i
__global__ void pack_b_frag(const float* __restrict__ W, int K,
                            unsigned short* __restrict__ dst,
                            int nt_off, int NT_total, int n_total) {
  int idx = blockIdx.x * blockDim.x + threadIdx.x;
  if (idx >= n_total) return;                 // n_total = KT*8*512
  int i    = idx & 15;
  int lane = (idx >> 4) & 31;
  int rest = idx >> 9;
  int nt   = rest & 7;                        // 8 local n-tiles (128 cols)
  int kt   = rest >> 3;
  int ksrc = kt * 32 + ((lane < 16) ? 0 : 16) + i;
  int col  = nt * 16 + (lane & 15);
  unsigned short v = 0;
  if (ksrc < K) v = f2bf(W[(size_t)ksrc * 128 + col]);
  dst[(((size_t)kt * NT_total + nt_off + nt) * 32 + lane) * 16 + i] = v;
}

// ---------------------------------------------------------------- WMMA GEMM
// OUT[M x Ncols] = A_bf16[M x Kpad] @ Wpacked + bias.
// Block: 4 waves share one 16-row A tile (double-buffered through LDS);
// each wave computes 4 consecutive 16-col tiles -> 4 WMMAs per A fragment.
__global__ __launch_bounds__(128) void wmma_gemm_bf16(
    const unsigned short* __restrict__ A, int Kpad,
    const unsigned short* __restrict__ Wp, const float* __restrict__ bias,
    float* __restrict__ OUT, int Ncols) {
  __shared__ unsigned short tileA[2][16 * 32];
  const int lane    = threadIdx.x & 31;
  const int wave    = threadIdx.x >> 5;
  const int rowBase = blockIdx.x * 16;
  const int NT      = Ncols >> 4;
  const int ntBase  = blockIdx.y * 16 + wave * 4;   // 4 n-tiles per wave
  const int KT      = Kpad >> 5;

  v8f acc[4] = {{}, {}, {}, {}};
  const int mloc = lane & 15;
  const int off  = (lane < 16) ? 0 : 16;

  // cooperative A-tile stage: 128 threads x 4 halves (8B each)
  const int str = threadIdx.x >> 3;          // row 0..15
  const int stk = (threadIdx.x & 7) * 4;     // k 0..28
  const unsigned short* Arow = A + (size_t)(rowBase + str) * Kpad + stk;

  *(uint2*)(&tileA[0][str * 32 + stk]) = *(const uint2*)(Arow);
  __syncthreads();

  for (int kt = 0; kt < KT; ++kt) {
    v16bf a;
    ((float4*)&a)[0] = *(const float4*)(&tileA[kt & 1][mloc * 32 + off]);
    ((float4*)&a)[1] = *(const float4*)(&tileA[kt & 1][mloc * 32 + off + 8]);
    if (kt + 1 < KT)   // prefetch next A tile into the other LDS buffer
      *(uint2*)(&tileA[(kt + 1) & 1][str * 32 + stk]) =
          *(const uint2*)(Arow + (size_t)(kt + 1) * 32);
#pragma unroll
    for (int j = 0; j < 4; ++j) {
      v16bf b;
      const unsigned short* bp =
          Wp + (((size_t)kt * NT + ntBase + j) * 32 + lane) * 16;
      ((float4*)&b)[0] = ((const float4*)bp)[0];
      ((float4*)&b)[1] = ((const float4*)bp)[1];
      acc[j] = __builtin_amdgcn_wmma_f32_16x16x32_bf16(false, a, false, b,
                                                       (short)0, acc[j], false,
                                                       false);
    }
    __syncthreads();
  }

  const int rofs = (lane < 16) ? 0 : 8;
#pragma unroll
  for (int j = 0; j < 4; ++j) {
    const int ncol = (ntBase + j) * 16 + mloc;
    const float bv = bias[ncol];
#pragma unroll
    for (int r = 0; r < 8; ++r)
      OUT[(size_t)(rowBase + r + rofs) * Ncols + ncol] = acc[j][r] + bv;
  }
}

// ---------------------------------------------------------------- edge passes
// qkvs layout per node: [0:128)=q  [128:256)=k  [256:384)=v  [384:512)=skip
__global__ __launch_bounds__(256) void edge_logits_max(
    const int* __restrict__ src, const int* __restrict__ dst,
    const float* __restrict__ eattr, const float* __restrict__ We,
    const float* __restrict__ qkvs, float* __restrict__ logits,
    float* __restrict__ mbuf, int E) {
  const int lane = threadIdx.x & 31;
  const int e = (blockIdx.x * blockDim.x + threadIdx.x) >> 5;
  if (e >= E) return;
  const int s = src[e], d = dst[e];
  const int f = lane * 4;
  const float4 q = *(const float4*)(qkvs + (size_t)d * 512 + f);
  const float4 k = *(const float4*)(qkvs + (size_t)s * 512 + 128 + f);
  float4 ep = make_float4(0.f, 0.f, 0.f, 0.f);
#pragma unroll
  for (int j = 0; j < 10; ++j) {
    const float aj = eattr[(size_t)e * 10 + j];
    const float4 w = *(const float4*)(We + j * 128 + f);
    ep.x = fmaf(aj, w.x, ep.x); ep.y = fmaf(aj, w.y, ep.y);
    ep.z = fmaf(aj, w.z, ep.z); ep.w = fmaf(aj, w.w, ep.w);
  }
  float p = q.x * (k.x + ep.x) + q.y * (k.y + ep.y) +
            q.z * (k.z + ep.z) + q.w * (k.w + ep.w);
#pragma unroll
  for (int o = 16; o >= 1; o >>= 1) p += __shfl_xor(p, o, 32);
  if (lane == 0) {
    const float lg = p * 0.08838834764831843f;   // 1/sqrt(128)
    logits[e] = lg;
    atomicMaxF(&mbuf[d], lg);
  }
}

__global__ void edge_exp_sum(const int* __restrict__ dst,
                             const float* __restrict__ logits,
                             const float* __restrict__ mbuf,
                             float* __restrict__ aexp, float* __restrict__ ssum,
                             int E) {
  int e = blockIdx.x * blockDim.x + threadIdx.x;
  if (e >= E) return;
  int d = dst[e];
  float av = __expf(logits[e] - mbuf[d]);
  aexp[e] = av;
  atomicAdd(&ssum[d], av);
}

__global__ __launch_bounds__(256) void edge_scatter(
    const int* __restrict__ src, const int* __restrict__ dst,
    const float* __restrict__ eattr, const float* __restrict__ We,
    const float* __restrict__ qkvs, const float* __restrict__ aexp,
    const float* __restrict__ ssum, float* __restrict__ agg, int E) {
  const int lane = threadIdx.x & 31;
  const int e = (blockIdx.x * blockDim.x + threadIdx.x) >> 5;
  if (e >= E) return;
  const int s = src[e], d = dst[e];
  const float coeff = aexp[e] / (ssum[d] + 1e-16f);
  const int f = lane * 4;
  const float4 v = *(const float4*)(qkvs + (size_t)s * 512 + 256 + f);
  float4 ep = make_float4(0.f, 0.f, 0.f, 0.f);
#pragma unroll
  for (int j = 0; j < 10; ++j) {
    const float aj = eattr[(size_t)e * 10 + j];
    const float4 w = *(const float4*)(We + j * 128 + f);
    ep.x = fmaf(aj, w.x, ep.x); ep.y = fmaf(aj, w.y, ep.y);
    ep.z = fmaf(aj, w.z, ep.z); ep.w = fmaf(aj, w.w, ep.w);
  }
  float* ag = agg + (size_t)d * 128 + f;
  atomicAdd(ag + 0, coeff * (v.x + ep.x));
  atomicAdd(ag + 1, coeff * (v.y + ep.y));
  atomicAdd(ag + 2, coeff * (v.z + ep.z));
  atomicAdd(ag + 3, coeff * (v.w + ep.w));
}

// h = relu(agg + skip); write fp32 (for pooling) and bf16 (next layer A)
__global__ void finalize_relu(const float* __restrict__ agg,
                              const float* __restrict__ qkvs,
                              float* __restrict__ h32,
                              unsigned short* __restrict__ xb, int n_total) {
  int idx = blockIdx.x * blockDim.x + threadIdx.x;
  if (idx >= n_total) return;
  int n = idx >> 7, dd = idx & 127;
  float hv = agg[idx] + qkvs[(size_t)n * 512 + 384 + dd];
  hv = fmaxf(hv, 0.f);
  h32[idx] = hv;
  xb[idx] = f2bf(hv);
}

// ---------------------------------------------------------------- pooling + head
__global__ void count_nodes(const int* __restrict__ batchs, float* __restrict__ cnt, int n) {
  int i = blockIdx.x * blockDim.x + threadIdx.x;
  if (i < n) atomicAdd(&cnt[batchs[i]], 1.0f);
}

__global__ void pool_sum(const float* __restrict__ h32, const int* __restrict__ batchs,
                         float* __restrict__ xc, int coloff, int n_total) {
  int idx = blockIdx.x * blockDim.x + threadIdx.x;
  if (idx >= n_total) return;
  int n = idx >> 7, dd = idx & 127;
  atomicAdd(&xc[(size_t)batchs[n] * 256 + coloff + dd], h32[idx]);
}

__global__ void pool_div(float* xc, const float* __restrict__ cnt, int n_total) {
  int idx = blockIdx.x * blockDim.x + threadIdx.x;
  if (idx >= n_total) return;
  xc[idx] /= fmaxf(cnt[idx >> 8], 1.0f);
}

__global__ void linear_k(const float* __restrict__ X, const float* __restrict__ W,
                         const float* __restrict__ b, float* __restrict__ Y,
                         int M, int K, int Nc, int relu) {
  int idx = blockIdx.x * blockDim.x + threadIdx.x;
  if (idx >= M * Nc) return;
  int g = idx / Nc, o = idx - g * Nc;
  float acc = b[o];
  for (int k = 0; k < K; ++k) acc = fmaf(X[(size_t)g * K + k], W[(size_t)k * Nc + o], acc);
  Y[idx] = relu ? fmaxf(acc, 0.f) : acc;
}

__global__ void softmax2_k(const float* __restrict__ Y, float* __restrict__ out, int M) {
  int g = blockIdx.x * blockDim.x + threadIdx.x;
  if (g >= M) return;
  float a0 = Y[2 * g], a1 = Y[2 * g + 1];
  float mx = fmaxf(a0, a1);
  float e0 = __expf(a0 - mx), e1 = __expf(a1 - mx);
  float s = e0 + e1;
  out[2 * g] = e0 / s;
  out[2 * g + 1] = e1 / s;
}

// ---------------------------------------------------------------- host driver
// conv leaf order (jax pytree alphabetical): We Wk Wq Ws Wv bk bq bs bv
enum { LWe = 0, LWk, LWq, LWs, LWv, Lbk, Lbq, Lbs, Lbv };

struct ConvL {
  const float* lf[9];
  int fin, Kpad, KT;
  unsigned short* wp;   // packed bf16 weights, NT=32
  float* bc;            // bias concat [512]: bq|bk|bv|bs
};

extern "C" void kernel_launch(void* const* d_in, const int* in_sizes, int n_in,
                              void* d_out, int out_size, void* d_ws, size_t ws_size,
                              hipStream_t stream) {
  (void)in_sizes; (void)n_in; (void)out_size; (void)ws_size;
  const float* x      = (const float*)d_in[0];
  const int*   ei     = (const int*)d_in[1];
  const float* ea     = (const float*)d_in[2];
  const int*   batchs = (const int*)d_in[3];
  const float* xa     = (const float*)d_in[4];
  const int*   eia    = (const int*)d_in[5];
  const float* eaa    = (const float*)d_in[6];

  ConvL holo[4], apo[4];
  auto fill_conv = [&](ConvL& c, int base, int fin) {
    for (int i = 0; i < 9; ++i) c.lf[i] = (const float*)d_in[base + i];
    c.fin = fin;
    c.Kpad = (fin + 31) & ~31;
    c.KT = c.Kpad >> 5;
  };
  fill_conv(holo[0], 7, 54);                               // c1
  fill_conv(apo[0], 16, 20);                               // c2
  for (int i = 0; i < 3; ++i) fill_conv(holo[1 + i], 25 + 9 * i, 128);  // cs
  for (int i = 0; i < 3; ++i) fill_conv(apo[1 + i], 52 + 9 * i, 128);   // cs2
  const float* fcW   = (const float*)d_in[79];
  const float* fcb   = (const float*)d_in[80];
  const float* linlW = (const float*)d_in[81];
  const float* linlb = (const float*)d_in[82];
  const float* lin2W = (const float*)d_in[83];
  const float* lin2b = (const float*)d_in[84];

  // workspace carve-up
  char* wsp = (char*)d_ws;
  auto alloc = [&](size_t bytes) -> void* {
    void* p = (void*)wsp;
    wsp += (bytes + 255) & ~(size_t)255;
    return p;
  };
  float* qkvs          = (float*)alloc((size_t)NNODES * 512 * 4);
  float* agg           = (float*)alloc((size_t)NNODES * 128 * 4);
  float* h32           = (float*)alloc((size_t)NNODES * 128 * 4);
  unsigned short* xb   = (unsigned short*)alloc((size_t)NNODES * 128 * 2);
  float* logits        = (float*)alloc((size_t)NEDGES * 4);
  float* aexp          = (float*)alloc((size_t)NEDGES * 4);
  float* mbuf          = (float*)alloc((size_t)NNODES * 4);
  float* ssum          = (float*)alloc((size_t)NNODES * 4);
  float* xc            = (float*)alloc((size_t)GG * 256 * 4);
  float* y1            = (float*)alloc((size_t)GG * 256 * 4);
  float* y2            = (float*)alloc((size_t)GG * 128 * 4);
  float* y3            = (float*)alloc((size_t)GG * 2 * 4);
  float* cnt           = (float*)alloc((size_t)GG * 4);
  for (int l = 0; l < 4; ++l) {
    holo[l].wp = (unsigned short*)alloc((size_t)holo[l].Kpad * 512 * 2);
    holo[l].bc = (float*)alloc(512 * 4);
    apo[l].wp  = (unsigned short*)alloc((size_t)apo[l].Kpad * 512 * 2);
    apo[l].bc  = (float*)alloc(512 * 4);
  }

  auto cdiv = [](int a, int b) { return (a + b - 1) / b; };

  // ---- pack all conv weights / biases (bf16 B-fragment order)
  auto pack_layer = [&](ConvL& c) {
    const int nW[4] = {LWq, LWk, LWv, LWs};
    const int nB[4] = {Lbq, Lbk, Lbv, Lbs};
    int ntot = c.KT * 8 * 512;
    for (int p = 0; p < 4; ++p) {
      pack_b_frag<<<cdiv(ntot, 256), 256, 0, stream>>>(c.lf[nW[p]], c.fin, c.wp,
                                                       p * 8, 32, ntot);
      copy_f32<<<1, 128, 0, stream>>>(c.lf[nB[p]], c.bc + p * 128, 128);
    }
  };
  for (int l = 0; l < 4; ++l) { pack_layer(holo[l]); pack_layer(apo[l]); }

  // ---- node counts (shared by both branches) and pooled-feature zero
  fill_f32<<<1, GG, 0, stream>>>(cnt, 0.f, GG);
  fill_f32<<<cdiv(GG * 256, 256), 256, 0, stream>>>(xc, 0.f, GG * 256);
  count_nodes<<<cdiv(NNODES, 256), 256, 0, stream>>>(batchs, cnt, NNODES);

  const int nh = NNODES * 128;               // elements of h / agg
  auto run_branch = [&](const float* xin, ConvL* L, const int* eidx,
                        const float* eattr, int coloff) {
    const int* esrc = eidx;
    const int* edst = eidx + NEDGES;
    // layer-0 input conversion
    cvt_pad<<<cdiv(NNODES * L[0].Kpad, 256), 256, 0, stream>>>(
        xin, xb, L[0].fin, L[0].Kpad, NNODES * L[0].Kpad);
    for (int l = 0; l < 4; ++l) {
      ConvL& c = L[l];
      // projections: qkvs = xb @ [Wq|Wk|Wv|Ws] + bias  (WMMA, 4 tiles/wave)
      wmma_gemm_bf16<<<dim3(NNODES / 16, 2), 128, 0, stream>>>(
          xb, c.Kpad, c.wp, c.bc, qkvs, 512);
      // attention accumulators
      fill_f32<<<cdiv(NNODES, 256), 256, 0, stream>>>(mbuf, -INFINITY, NNODES);
      fill_f32<<<cdiv(NNODES, 256), 256, 0, stream>>>(ssum, 0.f, NNODES);
      fill_f32<<<cdiv(nh, 256), 256, 0, stream>>>(agg, 0.f, nh);
      // segment softmax, three passes
      edge_logits_max<<<NEDGES / 8, 256, 0, stream>>>(esrc, edst, eattr, c.lf[LWe],
                                                      qkvs, logits, mbuf, NEDGES);
      edge_exp_sum<<<cdiv(NEDGES, 256), 256, 0, stream>>>(edst, logits, mbuf, aexp,
                                                          ssum, NEDGES);
      edge_scatter<<<NEDGES / 8, 256, 0, stream>>>(esrc, edst, eattr, c.lf[LWe],
                                                   qkvs, aexp, ssum, agg, NEDGES);
      // h = relu(agg + skip); also produce next layer's bf16 input
      finalize_relu<<<cdiv(nh, 256), 256, 0, stream>>>(agg, qkvs, h32, xb, nh);
    }
    pool_sum<<<cdiv(nh, 256), 256, 0, stream>>>(h32, batchs, xc, coloff, nh);
  };

  run_branch(x, holo, ei, ea, 0);
  run_branch(xa, apo, eia, eaa, 128);

  pool_div<<<cdiv(GG * 256, 256), 256, 0, stream>>>(xc, cnt, GG * 256);

  // ---- MLP head + softmax
  linear_k<<<cdiv(GG * 256, 256), 256, 0, stream>>>(xc, linlW, linlb, y1, GG, 256, 256, 1);
  linear_k<<<cdiv(GG * 128, 256), 256, 0, stream>>>(y1, lin2W, lin2b, y2, GG, 256, 128, 1);
  linear_k<<<1, 128, 0, stream>>>(y2, fcW, fcb, y3, GG, 128, 2, 0);
  softmax2_k<<<1, GG, 0, stream>>>(y3, (float*)d_out, GG);
}
